// HeteroGraphEncoder_6047313952839
// MI455X (gfx1250) — matrix operand order repaired and checked by reading, not compiled, not run on hardware
//
#include <hip/hip_runtime.h>
#include <hip/hip_bf16.h>

typedef __attribute__((ext_vector_type(16))) _Float16 v16h;
typedef __attribute__((ext_vector_type(8)))  _Float16 v8h;
typedef __attribute__((ext_vector_type(8)))  float    v8f;

#define HIDDIM 128
#define NHEADS 4

// ---------------- helpers ----------------
__device__ __forceinline__ float atomicMaxF(float* addr, float val) {
  // sign-split trick: monotone in int space for >=0, reversed for <0
  if (val >= 0.0f)
    return __int_as_float(atomicMax((int*)addr, __float_as_int(val)));
  else
    return __uint_as_float(atomicMin((unsigned int*)addr, __float_as_uint(val)));
}

// ---------------- elementwise / format kernels ----------------
__global__ void k_fill(float* __restrict__ p, float v, long long n) {
  long long stride = (long long)gridDim.x * blockDim.x;
  for (long long i = (long long)blockIdx.x * blockDim.x + threadIdx.x; i < n; i += stride)
    p[i] = v;
}

__global__ void k_cvt_f16(const float* __restrict__ src, _Float16* __restrict__ dst, long long n) {
  long long stride = (long long)gridDim.x * blockDim.x;
  for (long long i = (long long)blockIdx.x * blockDim.x + threadIdx.x; i < n; i += stride)
    dst[i] = (_Float16)src[i];
}

// WT[n*K + k] = W[k*N + n]  (f32 -> f16 transpose of a small weight)
__global__ void k_transpose_f16(const float* __restrict__ W, _Float16* __restrict__ WT,
                                int K, int N) {
  long long tot = (long long)K * N;
  long long stride = (long long)gridDim.x * blockDim.x;
  for (long long i = (long long)blockIdx.x * blockDim.x + threadIdx.x; i < tot; i += stride) {
    int k = (int)(i / N), c = (int)(i % N);
    WT[(long long)c * K + k] = (_Float16)W[i];
  }
}

// ---------------- WMMA GEMM: C[M,N] = A[M,128] * B[128,N], BT is [N,128] f16 ----------------
// one wave -> one 16x16 tile; 4 waves per block walk M; K=128 unrolled into 4 wmma ops.
// N is a template constant so the epilogue stores use immediate offsets.
template <int N>
__global__ void k_wmma_gemm(const _Float16* __restrict__ A, const _Float16* __restrict__ BT,
                            float* __restrict__ C, int M) {
  const int lane = threadIdx.x & 31;
  const int wave = threadIdx.x >> 5;
  const int m0 = (blockIdx.x * 4 + wave) * 16;
  const int n0 = blockIdx.y * 16;
  if (m0 >= M) return;                      // wave-uniform exit, EXEC stays all-1 for WMMA
  const int half = lane >> 4;               // 0: lanes 0-15, 1: lanes 16-31
  const int r    = lane & 15;
  int rA = m0 + r; rA = (rA < M) ? rA : (M - 1);   // arithmetic clamp, no divergence
  const _Float16* arow = A  + (size_t)rA * HIDDIM + half * 8;        // A-frag K-base (0 or 8)
  const _Float16* brow = BT + (size_t)(n0 + r) * HIDDIM + half * 8;  // B-frag K-base (0 or 8)

  v8f acc = {};
#pragma unroll
  for (int kc = 0; kc < HIDDIM; kc += 32) {
    v8h alo = *(const v8h*)(arow + kc);        // K = kc + base + 0..7
    v8h ahi = *(const v8h*)(arow + kc + 16);   // K = kc + base + 16..23
    v8h blo = *(const v8h*)(brow + kc);        // K = kc + base + 0..7
    v8h bhi = *(const v8h*)(brow + kc + 16);   // K = kc + base + 16..23
    v16h a, b;
#pragma unroll
    for (int j = 0; j < 8; ++j) {
      a[j] = alo[j]; a[8 + j] = ahi[j];
      b[j] = blo[j]; b[8 + j] = bhi[j];
    }
    acc = __builtin_amdgcn_wmma_f32_16x16x32_f16(false, a, false, b,
                                                 (short)0, acc, false, false);
  }
  // D layout: lane n = lane&15 ; rows m0 + half*8 + j , j=0..7 in acc[j]
  const int mB = m0 + half * 8;
  float* crow = C + (size_t)mB * N + (n0 + r);
  if (m0 + 16 <= M) {                       // full tile: unguarded, immediate offsets
#pragma unroll
    for (int j = 0; j < 8; ++j) crow[j * N] = acc[j];
  } else {                                  // rare ragged tail tile
#pragma unroll
    for (int j = 0; j < 8; ++j)
      if (mB + j < M) crow[j * N] = acc[j];
  }
}

// ---------------- GCN pieces ----------------
__global__ void k_deg(const int* __restrict__ dst, int E, float* __restrict__ deg) {
  long long stride = (long long)gridDim.x * blockDim.x;
  for (long long e = (long long)blockIdx.x * blockDim.x + threadIdx.x; e < E; e += stride)
    atomicAdd(&deg[dst[e]], 1.0f);
}

__global__ void k_rsqrt(const float* __restrict__ deg, float* __restrict__ dis, int n) {
  int i = blockIdx.x * blockDim.x + threadIdx.x;
  if (i < n) dis[i] = (deg[i] > 0.0f) ? rsqrtf(deg[i]) : 0.0f;
}

// O[i,c] = H[i,c]*dis[i]^2 + bias[c]   (self-loop term + bias)
__global__ void k_gcn_self(const float* __restrict__ H, const float* __restrict__ dis,
                           const float* __restrict__ bias, float* __restrict__ O, int n) {
  long long tot = (long long)n * HIDDIM;
  long long stride = (long long)gridDim.x * blockDim.x;
  for (long long i = (long long)blockIdx.x * blockDim.x + threadIdx.x; i < tot; i += stride) {
    int node = (int)(i >> 7), c = (int)(i & 127);
    float d = dis[node];
    O[i] = H[i] * d * d + bias[c];
  }
}

// 4 channels per thread: b128 loads of H, edge indices read once per 4 channels.
__global__ void k_gcn_edge(const int* __restrict__ src, const int* __restrict__ dst, int E,
                           const float* __restrict__ dis, const float* __restrict__ H,
                           float* __restrict__ O) {
  long long tot = (long long)E * (HIDDIM / 4);
  long long stride = (long long)gridDim.x * blockDim.x;
  for (long long i = (long long)blockIdx.x * blockDim.x + threadIdx.x; i < tot; i += stride) {
    int e = (int)(i >> 5), q = (int)(i & 31);      // q: group of 4 channels
    int s = src[e], d = dst[e];
    float nrm = dis[s] * dis[d];
    float4 hv = ((const float4*)(H + (size_t)s * HIDDIM))[q];
    float* o = O + (size_t)d * HIDDIM + q * 4;
    atomicAdd(o + 0, hv.x * nrm);
    atomicAdd(o + 1, hv.y * nrm);
    atomicAdd(o + 2, hv.z * nrm);
    atomicAdd(o + 3, hv.w * nrm);
  }
}

__global__ void k_relu_cvt(float* __restrict__ B, _Float16* __restrict__ X16,
                           long long n, int doCvt) {
  long long stride = (long long)gridDim.x * blockDim.x;
  for (long long i = (long long)blockIdx.x * blockDim.x + threadIdx.x; i < n; i += stride) {
    float v = B[i]; v = v > 0.0f ? v : 0.0f;
    B[i] = v;
    if (doCvt) X16[i] = (_Float16)v;
  }
}

// ---------------- GAT pieces ----------------
// a_s[n,h] = sum_c HS[n, h*128+c] * att_s[h,c]
__global__ void k_as(const float* __restrict__ HS, const float* __restrict__ att_s,
                     float* __restrict__ a_s, int Ns) {
  long long tot = (long long)Ns * NHEADS;
  long long stride = (long long)gridDim.x * blockDim.x;
  for (long long i = (long long)blockIdx.x * blockDim.x + threadIdx.x; i < tot; i += stride) {
    int n = (int)(i >> 2), h = (int)(i & 3);
    const float* row = HS + (size_t)n * (NHEADS * HIDDIM) + h * HIDDIM;
    const float* a   = att_s + h * HIDDIM;
    float s = 0.0f;
#pragma unroll 8
    for (int c = 0; c < HIDDIM; ++c) s += row[c] * a[c];
    a_s[i] = s;
  }
}

// v[h,k] = sum_c W[k, h*128+c] * att_d[h,c]   (folds dst-attention through W)
__global__ void k_v(const float* __restrict__ W, const float* __restrict__ att_d,
                    float* __restrict__ v) {
  int i = blockIdx.x * blockDim.x + threadIdx.x;
  if (i >= NHEADS * HIDDIM) return;
  int h = i >> 7, k = i & 127;
  float s = 0.0f;
#pragma unroll 8
  for (int c = 0; c < HIDDIM; ++c)
    s += W[(size_t)k * (NHEADS * HIDDIM) + h * HIDDIM + c] * att_d[h * HIDDIM + c];
  v[i] = s;
}

// a_d[n,h] = sum_k X[n,k] * v[h,k]
__global__ void k_ad(const float* __restrict__ X, const float* __restrict__ v,
                     float* __restrict__ a_d, int n) {
  long long tot = (long long)n * NHEADS;
  long long stride = (long long)gridDim.x * blockDim.x;
  for (long long i = (long long)blockIdx.x * blockDim.x + threadIdx.x; i < tot; i += stride) {
    int node = (int)(i >> 2), h = (int)(i & 3);
    const float* xr = X + (size_t)node * HIDDIM;
    const float* vr = v + h * HIDDIM;
    float s = 0.0f;
#pragma unroll 8
    for (int k = 0; k < HIDDIM; ++k) s += xr[k] * vr[k];
    a_d[i] = s;
  }
}

__device__ __forceinline__ float edge_logit(const int* src, const int* dst,
                                            const float* a_s, const float* a_d,
                                            int e, int h, int& d) {
  int s = src[e]; d = dst[e];
  float x = a_s[s * NHEADS + h] + a_d[d * NHEADS + h];
  return x > 0.0f ? x : 0.2f * x;   // leaky_relu 0.2
}

__global__ void k_edge_max(const int* __restrict__ src, const int* __restrict__ dst, int E,
                           const float* __restrict__ a_s, const float* __restrict__ a_d,
                           float* __restrict__ m) {
  long long tot = (long long)E * NHEADS;
  long long stride = (long long)gridDim.x * blockDim.x;
  for (long long i = (long long)blockIdx.x * blockDim.x + threadIdx.x; i < tot; i += stride) {
    int e = (int)(i >> 2), h = (int)(i & 3), d;
    float x = edge_logit(src, dst, a_s, a_d, e, h, d);
    atomicMaxF(&m[d * NHEADS + h], x);
  }
}

__global__ void k_edge_expsum(const int* __restrict__ src, const int* __restrict__ dst, int E,
                              const float* __restrict__ a_s, const float* __restrict__ a_d,
                              const float* __restrict__ m, float* __restrict__ denom) {
  long long tot = (long long)E * NHEADS;
  long long stride = (long long)gridDim.x * blockDim.x;
  for (long long i = (long long)blockIdx.x * blockDim.x + threadIdx.x; i < tot; i += stride) {
    int e = (int)(i >> 2), h = (int)(i & 3), d;
    float x = edge_logit(src, dst, a_s, a_d, e, h, d);
    atomicAdd(&denom[d * NHEADS + h], expf(x - m[d * NHEADS + h]));
  }
}

__global__ void k_edge_alpha(const int* __restrict__ src, const int* __restrict__ dst, int E,
                             const float* __restrict__ a_s, const float* __restrict__ a_d,
                             const float* __restrict__ m, const float* __restrict__ denom,
                             float* __restrict__ alpha) {
  long long tot = (long long)E * NHEADS;
  long long stride = (long long)gridDim.x * blockDim.x;
  for (long long i = (long long)blockIdx.x * blockDim.x + threadIdx.x; i < tot; i += stride) {
    int e = (int)(i >> 2), h = (int)(i & 3), d;
    float x = edge_logit(src, dst, a_s, a_d, e, h, d);
    alpha[i] = expf(x - m[d * NHEADS + h]) / (denom[d * NHEADS + h] + 1e-16f);
  }
}

// G[d,c] += 0.25 * sum_h alpha[e,h] * HS[s, h*128+c]; 4 channels per thread (b128 loads)
__global__ void k_edge_accum(const int* __restrict__ src, const int* __restrict__ dst, int E,
                             const float* __restrict__ alpha, const float* __restrict__ HS,
                             float* __restrict__ G) {
  long long tot = (long long)E * (HIDDIM / 4);
  long long stride = (long long)gridDim.x * blockDim.x;
  for (long long i = (long long)blockIdx.x * blockDim.x + threadIdx.x; i < tot; i += stride) {
    int e = (int)(i >> 5), q = (int)(i & 31);
    int s = src[e], d = dst[e];
    float4 al = *(const float4*)(alpha + (size_t)e * NHEADS);
    const float4* hbase = (const float4*)(HS + (size_t)s * (NHEADS * HIDDIM));
    float4 h0 = hbase[q];
    float4 h1 = hbase[32 + q];
    float4 h2 = hbase[64 + q];
    float4 h3 = hbase[96 + q];
    float ax = al.x * h0.x + al.y * h1.x + al.z * h2.x + al.w * h3.x;
    float ay = al.x * h0.y + al.y * h1.y + al.z * h2.y + al.w * h3.y;
    float az = al.x * h0.z + al.y * h1.z + al.z * h2.z + al.w * h3.z;
    float aw = al.x * h0.w + al.y * h1.w + al.z * h2.w + al.w * h3.w;
    float* g = G + (size_t)d * HIDDIM + q * 4;
    atomicAdd(g + 0, 0.25f * ax);
    atomicAdd(g + 1, 0.25f * ay);
    atomicAdd(g + 2, 0.25f * az);
    atomicAdd(g + 3, 0.25f * aw);
  }
}

__global__ void k_init_gout(const float* __restrict__ b1, const float* __restrict__ b2,
                            float* __restrict__ G, int n) {
  long long tot = (long long)n * HIDDIM;
  long long stride = (long long)gridDim.x * blockDim.x;
  for (long long i = (long long)blockIdx.x * blockDim.x + threadIdx.x; i < tot; i += stride) {
    int c = (int)(i & 127);
    G[i] = b1[c] + b2[c];
  }
}

// ---------------- final reductions ----------------
__global__ void k_colmean(const float* __restrict__ X, int rows, float scale,
                          float* __restrict__ out) {
  __shared__ float red[256];
  int c = blockIdx.x;
  float s = 0.0f;
  for (int r = threadIdx.x; r < rows; r += blockDim.x)
    s += X[(size_t)r * HIDDIM + c];
  red[threadIdx.x] = s;
  __syncthreads();
  for (int off = 128; off > 0; off >>= 1) {
    if ((int)threadIdx.x < off) red[threadIdx.x] += red[threadIdx.x + off];
    __syncthreads();
  }
  if (threadIdx.x == 0) out[c] = red[0] * scale;
}

__global__ void k_fuse(const float* __restrict__ cat, const float* __restrict__ fw,
                       const float* __restrict__ fb, float* __restrict__ out) {
  int j = threadIdx.x;
  if (j >= HIDDIM) return;
  float s = fb[j];
#pragma unroll 8
  for (int i = 0; i < 3 * HIDDIM; ++i) s += cat[i] * fw[(size_t)i * HIDDIM + j];
  out[j] = s;
}

// ---------------- host side ----------------
static inline int nblk(long long n) { return (int)((n + 255) / 256); }

struct GatBufs {
  _Float16* E16; _Float16* WT; float* HS; float* a_s; float* a_d;
  float* mbuf; float* denom; float* alpha; float* vsm; float* G; const float* X;
};

static void run_gat(const float* embed, int Ns, const int* src, const int* dst, int E,
                    const float* gw, const float* att_src, const float* att_dst,
                    int nGene, const GatBufs& b, hipStream_t st) {
  k_cvt_f16<<<nblk((long long)Ns * HIDDIM), 256, 0, st>>>(embed, b.E16, (long long)Ns * HIDDIM);
  k_transpose_f16<<<nblk(HIDDIM * NHEADS * HIDDIM), 256, 0, st>>>(gw, b.WT, HIDDIM, NHEADS * HIDDIM);
  dim3 g((Ns + 63) / 64, (NHEADS * HIDDIM) / 16);
  k_wmma_gemm<NHEADS * HIDDIM><<<g, 128, 0, st>>>(b.E16, b.WT, b.HS, Ns);
  k_as<<<nblk((long long)Ns * NHEADS), 256, 0, st>>>(b.HS, att_src, b.a_s, Ns);
  k_v<<<2, 256, 0, st>>>(gw, att_dst, b.vsm);
  k_ad<<<nblk((long long)nGene * NHEADS), 256, 0, st>>>(b.X, b.vsm, b.a_d, nGene);
  k_fill<<<nblk((long long)nGene * NHEADS), 256, 0, st>>>(b.mbuf, -1e30f, (long long)nGene * NHEADS);
  k_fill<<<nblk((long long)nGene * NHEADS), 256, 0, st>>>(b.denom, 0.0f, (long long)nGene * NHEADS);
  k_edge_max<<<nblk((long long)E * NHEADS), 256, 0, st>>>(src, dst, E, b.a_s, b.a_d, b.mbuf);
  k_edge_expsum<<<nblk((long long)E * NHEADS), 256, 0, st>>>(src, dst, E, b.a_s, b.a_d, b.mbuf, b.denom);
  k_edge_alpha<<<nblk((long long)E * NHEADS), 256, 0, st>>>(src, dst, E, b.a_s, b.a_d, b.mbuf, b.denom, b.alpha);
  k_edge_accum<<<nblk((long long)E * (HIDDIM / 4)), 256, 0, st>>>(src, dst, E, b.alpha, b.HS, b.G);
}

extern "C" void kernel_launch(void* const* d_in, const int* in_sizes, int n_in,
                              void* d_out, int out_size, void* d_ws, size_t ws_size,
                              hipStream_t stream) {
  const float* gene_nodes = (const float*)d_in[0];
  const int*   drug_e  = (const int*)d_in[1]; const int Ed = in_sizes[1] / 2;
  const int*   dis_e   = (const int*)d_in[2]; const int Ec = in_sizes[2] / 2;
  const int*   gene_e  = (const int*)d_in[3]; const int Eg = in_sizes[3] / 2;
  const float* gcn1_w = (const float*)d_in[4];
  const float* gcn1_b = (const float*)d_in[5];
  const float* gcn2_w = (const float*)d_in[6];
  const float* gcn2_b = (const float*)d_in[7];
  const float* drug_embed    = (const float*)d_in[8];  const int Nd = in_sizes[8] / HIDDIM;
  const float* disease_embed = (const float*)d_in[9];  const int Nc = in_sizes[9] / HIDDIM;
  const float* gat_d_w = (const float*)d_in[10];
  const float* gat_d_as = (const float*)d_in[11];
  const float* gat_d_ad = (const float*)d_in[12];
  const float* gat_d_b  = (const float*)d_in[13];
  const float* gat_c_w = (const float*)d_in[14];
  const float* gat_c_as = (const float*)d_in[15];
  const float* gat_c_ad = (const float*)d_in[16];
  const float* gat_c_b  = (const float*)d_in[17];
  const float* fuse_w = (const float*)d_in[18];
  const float* fuse_b = (const float*)d_in[19];
  const int nGene = in_sizes[0] / HIDDIM;
  const int Emax = (Ed > Ec) ? Ed : Ec;

  // ---- carve workspace ----
  char* w = (char*)d_ws;
  auto take = [&](size_t bytes) -> char* {
    char* p = w; w += (bytes + 255) & ~(size_t)255; return p;
  };
  _Float16* X16 = (_Float16*)take((size_t)nGene * HIDDIM * 2);
  float* A = (float*)take((size_t)nGene * HIDDIM * 4);   // GEMM output H
  float* B = (float*)take((size_t)nGene * HIDDIM * 4);   // aggregated x
  float* G = (float*)take((size_t)nGene * HIDDIM * 4);   // GAT gene output (both relations)
  _Float16* WT  = (_Float16*)take((size_t)(NHEADS * HIDDIM) * HIDDIM * 2);
  _Float16* E16 = (_Float16*)take((size_t)Nd * HIDDIM * 2);
  float* HS   = (float*)take((size_t)Nd * NHEADS * HIDDIM * 4);
  float* deg  = (float*)take((size_t)nGene * 4);
  float* dis  = (float*)take((size_t)nGene * 4);
  float* a_s  = (float*)take((size_t)Nd * NHEADS * 4);
  float* a_d  = (float*)take((size_t)nGene * NHEADS * 4);
  float* mbuf = (float*)take((size_t)nGene * NHEADS * 4);
  float* den  = (float*)take((size_t)nGene * NHEADS * 4);
  float* alpha = (float*)take((size_t)Emax * NHEADS * 4);
  float* vsm  = (float*)take(NHEADS * HIDDIM * 4);
  float* cat  = (float*)take(3 * HIDDIM * 4);

  const long long nGF = (long long)nGene * HIDDIM;
  dim3 gG((nGene + 63) / 64, HIDDIM / 16);

  // ---- degree / norm (deg init to 1.0 covers the self loop) ----
  k_fill<<<nblk(nGene), 256, 0, stream>>>(deg, 1.0f, nGene);
  k_deg<<<nblk(Eg), 256, 0, stream>>>(gene_e + Eg, Eg, deg);
  k_rsqrt<<<nblk(nGene), 256, 0, stream>>>(deg, dis, nGene);

  // ---- GCN layer 1 ----
  k_cvt_f16<<<nblk(nGF), 256, 0, stream>>>(gene_nodes, X16, nGF);
  k_transpose_f16<<<nblk(HIDDIM * HIDDIM), 256, 0, stream>>>(gcn1_w, WT, HIDDIM, HIDDIM);
  k_wmma_gemm<HIDDIM><<<gG, 128, 0, stream>>>(X16, WT, A, nGene);
  k_gcn_self<<<nblk(nGF), 256, 0, stream>>>(A, dis, gcn1_b, B, nGene);
  k_gcn_edge<<<nblk((long long)Eg * (HIDDIM / 4)), 256, 0, stream>>>(gene_e, gene_e + Eg, Eg, dis, A, B);
  k_relu_cvt<<<nblk(nGF), 256, 0, stream>>>(B, X16, nGF, 1);

  // ---- GCN layer 2 ----
  k_transpose_f16<<<nblk(HIDDIM * HIDDIM), 256, 0, stream>>>(gcn2_w, WT, HIDDIM, HIDDIM);
  k_wmma_gemm<HIDDIM><<<gG, 128, 0, stream>>>(X16, WT, A, nGene);
  k_gcn_self<<<nblk(nGF), 256, 0, stream>>>(A, dis, gcn2_b, B, nGene);
  k_gcn_edge<<<nblk((long long)Eg * (HIDDIM / 4)), 256, 0, stream>>>(gene_e, gene_e + Eg, Eg, dis, A, B);
  k_relu_cvt<<<nblk(nGF), 256, 0, stream>>>(B, X16, nGF, 0);   // x (f32) stays in B

  // ---- hetero GAT into G (G starts at sum of both biases; 0.5 folded at mean) ----
  k_init_gout<<<nblk(nGF), 256, 0, stream>>>(gat_d_b, gat_c_b, G, nGene);
  GatBufs gb{E16, WT, HS, a_s, a_d, mbuf, den, alpha, vsm, G, B};
  run_gat(drug_embed,    Nd, drug_e, drug_e + Ed, Ed, gat_d_w, gat_d_as, gat_d_ad, nGene, gb, stream);
  run_gat(disease_embed, Nc, dis_e,  dis_e + Ec,  Ec, gat_c_w, gat_c_as, gat_c_ad, nGene, gb, stream);

  // ---- means + fuse ----
  k_colmean<<<HIDDIM, 256, 0, stream>>>(G, nGene, 0.5f / (float)nGene, cat);          // bg (0.5 folded)
  k_colmean<<<HIDDIM, 256, 0, stream>>>(drug_embed, Nd, 1.0f / (float)Nd, cat + HIDDIM);
  k_colmean<<<HIDDIM, 256, 0, stream>>>(disease_embed, Nc, 1.0f / (float)Nc, cat + 2 * HIDDIM);
  k_fuse<<<1, HIDDIM, 0, stream>>>(cat, fuse_w, fuse_b, (float*)d_out);
}
